// LSTM_17772574671229
// MI455X (gfx1250) — matrix operand order repaired and checked by reading, not compiled, not run on hardware
//
#include <hip/hip_runtime.h>
#include <hip/hip_bf16.h>

typedef __attribute__((ext_vector_type(16))) __bf16 v16bf;
typedef __attribute__((ext_vector_type(8)))  float  v8f;

#define H 512
#define B 256
#define TSTEPS 256
#define NCLS 10

// ---------------------------------------------------------------------------
// Pack W_gh/W_ih/W_fh/W_oh (fp32 row-major 512x512 each) into a bf16 buffer
// pre-swizzled into WMMA A fragments, k contiguous per lane half (B uses the same):
//   dest index d = (((rowTile*16 + kTile)*32 + lane)*16 + e)
//   lanes 0-15  : M = lane,    K = kt*32 + e
//   lanes 16-31 : M = lane-16, K = kt*32 + 16 + e
// Gates stacked: rows 0-511 = g, 512-1023 = i, 1024-1535 = f, 1536-2047 = o.
// ---------------------------------------------------------------------------
__global__ void pack_weights(const float* __restrict__ Wg, const float* __restrict__ Wi,
                             const float* __restrict__ Wf, const float* __restrict__ Wo,
                             __bf16* __restrict__ packedA) {
    int d = blockIdx.x * blockDim.x + threadIdx.x;
    if (d >= 2048 * 512) return;
    int e     = d & 15;
    int lane  = (d >> 4) & 31;
    int kt    = (d >> 9) & 15;
    int rt    = d >> 13;                 // 0..127 row tiles over 2048 rows
    int row   = rt * 16 + (lane & 15);
    int khalf = lane >> 4;
    int k     = kt * 32 + khalf * 16 + e;
    int gate  = row >> 9;
    int r     = row & 511;
    const float* W = (gate == 0) ? Wg : (gate == 1) ? Wi : (gate == 2) ? Wf : Wo;
    packedA[d] = (__bf16)W[r * 512 + k];
}

// Zero h (both buffers, bf16, [B][H] transposed layout) and c (fp32, [H][B]).
__global__ void zero_state(__bf16* __restrict__ h0, __bf16* __restrict__ h1,
                           float* __restrict__ cSt) {
    int i = blockIdx.x * blockDim.x + threadIdx.x;
    if (i >= B * H) return;
    h0[i] = (__bf16)0.0f;
    h1[i] = (__bf16)0.0f;
    cSt[i] = 0.0f;
}

__device__ __forceinline__ float sigmoidf_(float v) {
    return 1.0f / (1.0f + __expf(-v));
}

// ---------------------------------------------------------------------------
// One LSTM time step, fully fused. Grid: 64 blocks x 128 threads (4 waves).
// All 4 waves of a block share the same h-row tile m, so the block's entire
// weight working set (4 gates x 16 K-slices x 1KB = 64KB, time-invariant) is
// staged into LDS once. The hot loop double-buffers BOTH the A fragments
// (from LDS) and the B fragments (from global) one K-slice ahead, so the
// 8 WMMAs of each slice cover the ds_load/global_load latency of the next.
//   hSrc: bf16 [B][H] (transposed h of previous step)  -> B operand rows
//   hDst: bf16 [B][H] new h (double buffered vs hSrc)
//   cSt : fp32 [H][B], owned exclusively per tile (no cross-wave races)
// ---------------------------------------------------------------------------
__global__ __launch_bounds__(128) void lstm_step(
    const __bf16* __restrict__ packedA,
    const __bf16* __restrict__ hSrc, __bf16* __restrict__ hDst,
    float* __restrict__ cSt, const float* __restrict__ x,
    const float* __restrict__ Wgx, const float* __restrict__ bg,
    const float* __restrict__ Wix, const float* __restrict__ bi,
    const float* __restrict__ Wfx, const float* __restrict__ bfv,
    const float* __restrict__ Wox, const float* __restrict__ bo,
    int t)
{
    // LDS A-fragment cache: index = ((gate*16 + kt)*32 + lane)*16 + e  (bf16)
    __shared__ __align__(16) __bf16 ldsA[4 * 16 * 32 * 16];   // 64 KB

    const int lane = threadIdx.x & 31;
    const int wave = threadIdx.x >> 5;
    const int tile = blockIdx.x * 4 + wave;   // 0..255
    const int m  = tile >> 3;                 // h-row tile (uniform per block)
    const int np = tile & 7;                  // column-pair id  0..7
    const size_t gStride = (size_t)512 * 512; // per-gate block in packedA

    // ---- stage this block's 64KB of A fragments into LDS ------------------
    {
        const int mBlk = blockIdx.x >> 1;     // == m, uniform
        #pragma unroll 4
        for (int it = 0; it < 32; ++it) {
            int c  = it * 128 + threadIdx.x;  // 16-byte chunk id, 0..4095
            int hh = c & 1;                   // which 8-element half
            int ln = (c >> 1) & 31;
            int kk = (c >> 6) & 15;
            int g  = c >> 10;
            const __bf16* src = packedA + (size_t)g * gStride
                              + (((size_t)(mBlk * 16 + kk) * 32 + ln) << 4) + hh * 8;
            uint4 v = *reinterpret_cast<const uint4*>(src);
            *reinterpret_cast<uint4*>(&ldsA[(((g * 16 + kk) * 32 + ln) << 4) + hh * 8]) = v;
        }
    }
    __syncthreads();

    const int nl   = lane & 15;
    const int half = lane >> 4;
    const int col0 = np * 32 + nl;            // batch column, first tile
    const int col1 = col0 + 16;               // batch column, second tile

    v8f aG0 = {}, aI0 = {}, aF0 = {}, aO0 = {};
    v8f aG1 = {}, aI1 = {}, aF1 = {}, aO1 = {};

    const __bf16* hRow0 = hSrc + (size_t)col0 * H + half * 16;
    const __bf16* hRow1 = hSrc + (size_t)col1 * H + half * 16;

    // LDS base for this lane; gate stride = 8192 elements, kt stride = 512.
    const __bf16* lBase = &ldsA[lane << 4];

    // ---- prologue: fragments for K-slice 0 --------------------------------
    v16bf fB0 = *reinterpret_cast<const v16bf*>(hRow0);
    v16bf fB1 = *reinterpret_cast<const v16bf*>(hRow1);
    v16bf fAG = *reinterpret_cast<const v16bf*>(lBase);
    v16bf fAI = *reinterpret_cast<const v16bf*>(lBase + 1 * 8192);
    v16bf fAF = *reinterpret_cast<const v16bf*>(lBase + 2 * 8192);
    v16bf fAO = *reinterpret_cast<const v16bf*>(lBase + 3 * 8192);

    #pragma unroll
    for (int kt = 0; kt < 16; ++kt) {
        // issue next K-slice's loads (A from LDS, B from global) first so
        // the 8 WMMAs below hide their latency
        const int ktn = (kt + 1 < 16) ? (kt + 1) : kt;
        v16bf nB0 = *reinterpret_cast<const v16bf*>(hRow0 + ktn * 32);
        v16bf nB1 = *reinterpret_cast<const v16bf*>(hRow1 + ktn * 32);
        const __bf16* lA = lBase + ktn * 512;
        v16bf nAG = *reinterpret_cast<const v16bf*>(lA);
        v16bf nAI = *reinterpret_cast<const v16bf*>(lA + 1 * 8192);
        v16bf nAF = *reinterpret_cast<const v16bf*>(lA + 2 * 8192);
        v16bf nAO = *reinterpret_cast<const v16bf*>(lA + 3 * 8192);

        aG0 = __builtin_amdgcn_wmma_f32_16x16x32_bf16(false, fAG, false, fB0, (short)0, aG0, false, false);
        aG1 = __builtin_amdgcn_wmma_f32_16x16x32_bf16(false, fAG, false, fB1, (short)0, aG1, false, false);
        aI0 = __builtin_amdgcn_wmma_f32_16x16x32_bf16(false, fAI, false, fB0, (short)0, aI0, false, false);
        aI1 = __builtin_amdgcn_wmma_f32_16x16x32_bf16(false, fAI, false, fB1, (short)0, aI1, false, false);
        aF0 = __builtin_amdgcn_wmma_f32_16x16x32_bf16(false, fAF, false, fB0, (short)0, aF0, false, false);
        aF1 = __builtin_amdgcn_wmma_f32_16x16x32_bf16(false, fAF, false, fB1, (short)0, aF1, false, false);
        aO0 = __builtin_amdgcn_wmma_f32_16x16x32_bf16(false, fAO, false, fB0, (short)0, aO0, false, false);
        aO1 = __builtin_amdgcn_wmma_f32_16x16x32_bf16(false, fAO, false, fB1, (short)0, aO1, false, false);

        fB0 = nB0; fB1 = nB1;
        fAG = nAG; fAI = nAI; fAF = nAF; fAO = nAO;
    }

    // Epilogue: C-fragment element i lives at (M = i + 8*half, N = nl).
    // Per-row vectors are contiguous over vi -> fetch as 32-byte v8f loads.
    const int rBase = m * 16 + half * 8;
    const v8f wgv = *reinterpret_cast<const v8f*>(Wgx + rBase);
    const v8f wiv = *reinterpret_cast<const v8f*>(Wix + rBase);
    const v8f wfv = *reinterpret_cast<const v8f*>(Wfx + rBase);
    const v8f wov = *reinterpret_cast<const v8f*>(Wox + rBase);
    const v8f bgv = *reinterpret_cast<const v8f*>(bg  + rBase);
    const v8f biv = *reinterpret_cast<const v8f*>(bi  + rBase);
    const v8f bfq = *reinterpret_cast<const v8f*>(bfv + rBase);
    const v8f bov = *reinterpret_cast<const v8f*>(bo  + rBase);

    const float xv0 = x[(size_t)col0 * TSTEPS + t];
    const float xv1 = x[(size_t)col1 * TSTEPS + t];
    #pragma unroll
    for (int vi = 0; vi < 8; ++vi) {
        const int row = rBase + vi;              // h index
        const float wg = wgv[vi], wi = wiv[vi], wf = wfv[vi], wo = wov[vi];
        const float cg = bgv[vi], ci = biv[vi], cf = bfq[vi], co = bov[vi];

        // column tile 0
        {
            float g  = tanhf(aG0[vi] + wg * xv0 + cg);
            float ig = sigmoidf_(aI0[vi] + wi * xv0 + ci);
            float fg = sigmoidf_(aF0[vi] + wf * xv0 + cf);
            float og = sigmoidf_(aO0[vi] + wo * xv0 + co);
            float cOld = cSt[(size_t)row * B + col0];
            float cNew = g * ig + cOld * fg;
            cSt[(size_t)row * B + col0] = cNew;
            hDst[(size_t)col0 * H + row] = (__bf16)(tanhf(cNew) * og);
        }
        // column tile 1
        {
            float g  = tanhf(aG1[vi] + wg * xv1 + cg);
            float ig = sigmoidf_(aI1[vi] + wi * xv1 + ci);
            float fg = sigmoidf_(aF1[vi] + wf * xv1 + cf);
            float og = sigmoidf_(aO1[vi] + wo * xv1 + co);
            float cOld = cSt[(size_t)row * B + col1];
            float cNew = g * ig + cOld * fg;
            cSt[(size_t)row * B + col1] = cNew;
            hDst[(size_t)col1 * H + row] = (__bf16)(tanhf(cNew) * og);
        }
    }
}

// ---------------------------------------------------------------------------
// Final projection p = W_ph @ h + bias_p, softmax over classes, output (B, C).
// One wave per batch element; butterfly reduction across the wave.
// ---------------------------------------------------------------------------
__global__ __launch_bounds__(32) void proj_softmax(
    const __bf16* __restrict__ hT, const float* __restrict__ Wph,
    const float* __restrict__ bp, float* __restrict__ out)
{
    const int b = blockIdx.x;
    const int lane = threadIdx.x;
    float acc[NCLS];
    #pragma unroll
    for (int c = 0; c < NCLS; ++c) acc[c] = 0.0f;

    for (int k = lane; k < H; k += 32) {
        float hv = (float)hT[(size_t)b * H + k];
        #pragma unroll
        for (int c = 0; c < NCLS; ++c) acc[c] += Wph[c * H + k] * hv;
    }
    #pragma unroll
    for (int c = 0; c < NCLS; ++c) {
        for (int off = 16; off > 0; off >>= 1)
            acc[c] += __shfl_xor(acc[c], off, 32);
    }
    if (lane == 0) {
        float v[NCLS], mx = -1e30f;
        #pragma unroll
        for (int c = 0; c < NCLS; ++c) { v[c] = acc[c] + bp[c]; mx = fmaxf(mx, v[c]); }
        float s = 0.0f;
        #pragma unroll
        for (int c = 0; c < NCLS; ++c) { v[c] = __expf(v[c] - mx); s += v[c]; }
        float inv = 1.0f / s;
        #pragma unroll
        for (int c = 0; c < NCLS; ++c) out[(size_t)b * NCLS + c] = v[c] * inv;
    }
}

extern "C" void kernel_launch(void* const* d_in, const int* in_sizes, int n_in,
                              void* d_out, int out_size, void* d_ws, size_t ws_size,
                              hipStream_t stream) {
    const float* x      = (const float*)d_in[0];
    const float* W_gx   = (const float*)d_in[1];
    const float* W_gh   = (const float*)d_in[2];
    const float* bias_g = (const float*)d_in[3];
    const float* W_ix   = (const float*)d_in[4];
    const float* W_ih   = (const float*)d_in[5];
    const float* bias_i = (const float*)d_in[6];
    const float* W_fx   = (const float*)d_in[7];
    const float* W_fh   = (const float*)d_in[8];
    const float* bias_f = (const float*)d_in[9];
    const float* W_ox   = (const float*)d_in[10];
    const float* W_oh   = (const float*)d_in[11];
    const float* bias_o = (const float*)d_in[12];
    const float* W_ph   = (const float*)d_in[13];
    const float* bias_p = (const float*)d_in[14];

    char* ws = (char*)d_ws;
    __bf16* packedA = (__bf16*)ws;                                   // 2 MB
    __bf16* hT0     = (__bf16*)(ws + 2048u * 512u * 2u);             // 256 KB
    __bf16* hT1     = (__bf16*)(ws + 2048u * 512u * 2u + B * H * 2u);
    float*  cSt     = (float*)(ws + 2048u * 512u * 2u + 2u * B * H * 2u);

    pack_weights<<<(2048 * 512) / 256, 256, 0, stream>>>(W_gh, W_ih, W_fh, W_oh, packedA);
    zero_state<<<(B * H) / 256, 256, 0, stream>>>(hT0, hT1, cSt);

    for (int t = 0; t < TSTEPS; ++t) {
        const __bf16* hSrc = (t & 1) ? hT1 : hT0;
        __bf16*       hDst = (t & 1) ? hT0 : hT1;
        lstm_step<<<64, 128, 0, stream>>>(packedA, hSrc, hDst, cSt, x,
                                          W_gx, bias_g, W_ix, bias_i,
                                          W_fx, bias_f, W_ox, bias_o, t);
    }
    // After t = 255 (odd), the final h lives in hT0.
    proj_softmax<<<B, 32, 0, stream>>>(hT0, W_ph, bias_p, (float*)d_out);
}
